// ViTVectorQuantizer_45595372814699
// MI455X (gfx1250) — matrix operand order repaired and checked by reading, not compiled, not run on hardware
//
#include <hip/hip_runtime.h>
#include <math.h>

typedef _Float16 half8  __attribute__((ext_vector_type(8)));
typedef _Float16 half16 __attribute__((ext_vector_type(16)));
typedef float    f32x8  __attribute__((ext_vector_type(8)));

#define N_Z   32768   // 32*1024 query rows
#define N_E   8192    // codebook entries
#define DIM   32
#define CHUNKS (N_E / 16)   // 512 16-code chunks over the codebook
#define ZQ_ELEMS (N_Z * DIM)        // 1048576 floats of z_q_out
#define LOSS_OFF ZQ_ELEMS           // then 1 float loss
#define IDX_OFF  (ZQ_ELEMS + 1)     // then 32768 idx-as-float
#define FINAL_BLOCKS 128
#define ROWS_PER_WAVE 32            // two 16-row A tiles per wave

// ---------------------------------------------------------------------------
// Kernel 1: normalize embedding -> en (f32 row-major) + ebt (f16, tiled
// transpose laid out exactly as the WMMA B operand wants it:
//   ebt[tile(=n>>4)][khalf][n&15][16 k in table order] -> lane loads 16 contiguous halves),
// and normalize z -> zf (f16 row-major, the WMMA A operand source).
// ---------------------------------------------------------------------------
__global__ void __launch_bounds__(256) vq_prep(const float* __restrict__ z,
                                               const float* __restrict__ emb,
                                               float* __restrict__ en,
                                               _Float16* __restrict__ ebt,
                                               _Float16* __restrict__ zf) {
  int t = blockIdx.x * blockDim.x + threadIdx.x;
  if (t < N_E) {
    const float* r = emb + t * DIM;
    float x[DIM];
    float s = 0.f;
#pragma unroll
    for (int d = 0; d < DIM; ++d) { x[d] = r[d]; s += x[d] * x[d]; }
    float inv = 1.0f / fmaxf(sqrtf(s), 1e-12f);
    int tile = t >> 4, col = t & 15;
#pragma unroll
    for (int d = 0; d < DIM; ++d) {
      float v = x[d] * inv;
      en[t * DIM + d] = v;
      ebt[tile * (32 * 16) + (((d >> 3) & 1) * 16 + col) * 16 + (d & 7) + ((d >> 4) << 3)] = (_Float16)v;
    }
  } else if (t < N_E + N_Z) {
    int n = t - N_E;
    const float* r = z + n * DIM;
    float x[DIM];
    float s = 0.f;
#pragma unroll
    for (int d = 0; d < DIM; ++d) { x[d] = r[d]; s += x[d] * x[d]; }
    float inv = 1.0f / fmaxf(sqrtf(s), 1e-12f);
#pragma unroll
    for (int d = 0; d < DIM; ++d) zf[n * DIM + d] = (_Float16)(x[d] * inv);
  }
}

// ---------------------------------------------------------------------------
// Kernel 2: fused GEMM + argmax.
//  * One wave32 owns TWO 16-row A tiles (resident in 16 VGPRs) -> two
//    independent v_wmma_f32_16x16x32_f16 per B-tile fetch (halves L2 traffic,
//    fills the WMMA->VALU hazard window with real work).
//  * B tile is double-buffered in registers: the load for chunk c+1 is issued
//    before chunk c is consumed, so s_wait_loadcnt lands a full iteration
//    after issue instead of immediately.
//  * Row scores land in one 16-lane half of one VGPR -> final argmax is a
//    4-step shfl_xor reduction with first-index tie-breaking.
// ---------------------------------------------------------------------------
__global__ void __launch_bounds__(256) vq_argmax(const _Float16* __restrict__ zf,
                                                 const _Float16* __restrict__ ebt,
                                                 int* __restrict__ idxOut) {
  const int lane  = threadIdx.x & 31;
  const int wave  = threadIdx.x >> 5;
  const int rowTile = (blockIdx.x * 8 + wave) * ROWS_PER_WAVE;
  const int m     = lane & 15;      // row within tile (A) / col within chunk (D)
  const int khalf = lane >> 4;      // which K-half this lane holds for A

  // A operands: lane holds row m, K = {8*khalf..8*khalf+7, 16+8*khalf..+7}
  half16 a0, a1;
  {
    const _Float16* r0 = zf + (size_t)(rowTile + m) * DIM;
    const _Float16* r1 = zf + (size_t)(rowTile + 16 + m) * DIM;
    half8 lo0 = *(const half8*)(r0 + 8 * khalf);
    half8 hi0 = *(const half8*)(r0 + 16 + 8 * khalf);
    half8 lo1 = *(const half8*)(r1 + 8 * khalf);
    half8 hi1 = *(const half8*)(r1 + 16 + 8 * khalf);
#pragma unroll
    for (int i = 0; i < 8; ++i) {
      a0[i] = lo0[i]; a0[i + 8] = hi0[i];
      a1[i] = lo1[i]; a1[i + 8] = hi1[i];
    }
  }

  float bv0[8], bv1[8];
  int   bi0[8], bi1[8];
#pragma unroll
  for (int r = 0; r < 8; ++r) {
    bv0[r] = -3.4e38f; bi0[r] = 0;
    bv1[r] = -3.4e38f; bi1[r] = 0;
  }

  // B operand: lane holds column lane&15, K-half lane>>4 (table order), 16 contiguous f16.
  const _Float16* bbase = ebt + lane * 16;

  auto process = [&](half16 b, int chunk) {
    f32x8 zero = {0.f, 0.f, 0.f, 0.f, 0.f, 0.f, 0.f, 0.f};
    f32x8 c0 = __builtin_amdgcn_wmma_f32_16x16x32_f16(
        false, a0, false, b, (short)0, zero, false, false);
    f32x8 c1 = __builtin_amdgcn_wmma_f32_16x16x32_f16(
        false, a1, false, b, (short)0, zero, false, false);
    int id = (chunk << 4) + m;
    // chunk index increases monotonically -> strict '>' keeps first max,
    // matching jnp.argmin first-tie semantics.
#pragma unroll
    for (int r = 0; r < 8; ++r) {
      if (c0[r] > bv0[r]) { bv0[r] = c0[r]; bi0[r] = id; }
      if (c1[r] > bv1[r]) { bv1[r] = c1[r]; bi1[r] = id; }
    }
  };

  // Software-pipelined sweep over the codebook with double-buffered B.
  half16 bcur = *(const half16*)(bbase);
  for (int chunk = 0; chunk < CHUNKS - 1; ++chunk) {
    half16 bnext = *(const half16*)(bbase + (size_t)(chunk + 1) * (32 * 16));
    if (chunk + 2 < CHUNKS)
      __builtin_prefetch(bbase + (size_t)(chunk + 2) * (32 * 16), 0, 1);
    process(bcur, chunk);
    bcur = bnext;
  }
  process(bcur, CHUNKS - 1);

  // Cross-lane argmax within each 16-lane half (rows r / r+8 of each tile).
  auto reduceWrite = [&](float* bv, int* bi, int rowBase) {
#pragma unroll
    for (int r = 0; r < 8; ++r) {
      float v = bv[r];
      int   id = bi[r];
#pragma unroll
      for (int off = 8; off >= 1; off >>= 1) {
        float ov = __shfl_xor(v, off, 32);
        int   oi = __shfl_xor(id, off, 32);
        if (ov > v || (ov == v && oi < id)) { v = ov; id = oi; }
      }
      if (m == 0) idxOut[rowBase + (khalf << 3) + r] = id;
    }
  };
  reduceWrite(bv0, bi0, rowTile);
  reduceWrite(bv1, bi1, rowTile + 16);
}

// ---------------------------------------------------------------------------
// Kernel 3: gather + straight-through output + loss partials (deterministic
// tree reduction per block, no float atomics).
// ---------------------------------------------------------------------------
__global__ void __launch_bounds__(256) vq_final(const float* __restrict__ z,
                                                const float* __restrict__ en,
                                                const int* __restrict__ idx,
                                                float* __restrict__ out,
                                                float* __restrict__ partials) {
  __shared__ float red[256];
  int n = blockIdx.x * blockDim.x + threadIdx.x;   // 0..32767
  float s = 0.f;

  const float* zr = z + (size_t)n * DIM;
  float x[DIM];
  float ss = 0.f;
#pragma unroll
  for (int d = 0; d < DIM; ++d) { x[d] = zr[d]; ss += x[d] * x[d]; }
  float inv = 1.0f / fmaxf(sqrtf(ss), 1e-12f);

  int k = idx[n];
  const float* q = en + (size_t)k * DIM;           // z_qnorm row
#pragma unroll
  for (int d = 0; d < DIM; ++d) {
    float qd = q[d];
    float zd = x[d];
    out[(size_t)n * DIM + d] = zd + (qd - zd);     // straight-through estimator
    float diff = qd - zd * inv;                    // z_qnorm - z_norm
    s += diff * diff;
  }
  out[IDX_OFF + n] = (float)k;

  red[threadIdx.x] = s;
  __syncthreads();
  for (int st = 128; st > 0; st >>= 1) {
    if (threadIdx.x < st) red[threadIdx.x] += red[threadIdx.x + st];
    __syncthreads();
  }
  if (threadIdx.x == 0) partials[blockIdx.x] = red[0];
}

__global__ void vq_loss(const float* __restrict__ partials, float* __restrict__ out) {
  if (threadIdx.x == 0) {
    float s = 0.f;
    for (int i = 0; i < FINAL_BLOCKS; ++i) s += partials[i];  // fixed order
    // loss = (BETA + 1) * mean((z_qnorm - z_norm)^2), mean over B*T*D
    out[LOSS_OFF] = (1.0f + 0.25f) * s / (float)(N_Z * DIM);
  }
}

// ---------------------------------------------------------------------------
extern "C" void kernel_launch(void* const* d_in, const int* in_sizes, int n_in,
                              void* d_out, int out_size, void* d_ws, size_t ws_size,
                              hipStream_t stream) {
  (void)in_sizes; (void)n_in; (void)out_size; (void)ws_size;

  const float* z   = (const float*)d_in[0];   // [32,1024,32]
  const float* emb = (const float*)d_in[1];   // [8192,32]
  float* out = (float*)d_out;

  // Workspace carve-up (all offsets are large powers of two -> aligned).
  float*    en  = (float*)d_ws;                         // 8192*32 f32   (1 MB)
  _Float16* ebt = (_Float16*)(en + N_E * DIM);          // 8192*32 f16   (512 KB)
  _Float16* zf  = ebt + N_E * DIM;                      // 32768*32 f16  (2 MB)
  int*      idx = (int*)(zf + (size_t)N_Z * DIM);       // 32768 i32
  float*    partials = (float*)(idx + N_Z);             // 128 f32

  vq_prep<<<(N_E + N_Z) / 256, 256, 0, stream>>>(z, emb, en, ebt, zf);
  vq_argmax<<<N_Z / (ROWS_PER_WAVE * 8), 256, 0, stream>>>(zf, ebt, idx); // 128 blocks
  vq_final<<<FINAL_BLOCKS, 256, 0, stream>>>(z, en, idx, out, partials);
  vq_loss<<<1, 64, 0, stream>>>(partials, out);
}
